// ANFISLayer_24807731102138
// MI455X (gfx1250) — matrix operand order repaired and checked by reading, not compiled, not run on hardware
//
#include <hip/hip_runtime.h>

typedef __attribute__((ext_vector_type(16))) _Float16 v16h;
typedef __attribute__((ext_vector_type(8)))  _Float16 v8h;
typedef __attribute__((ext_vector_type(4)))  _Float16 v4h;
typedef __attribute__((ext_vector_type(2)))  _Float16 h2;
typedef __attribute__((ext_vector_type(8)))  float    v8f;

// ---------------------------------------------------------------------------
// Pack consequents [4096 x 7] f32 into WMMA B-fragments (f16) in d_ws.
// Per K-chunk k (32 rules): 32 lanes x 16 halves contiguous.
//   value(k, lane, e) = Cpad[r, n],  n = lane & 15,
//   r = k*32 + (e < 8 ? 0 : 16) + (lane < 16 ? 0 : 8) + (e & 7)
//   Cpad[r, n] = n<7 ? consequents[r*7+n] : (n==7 ? 1.0f : 0.0f)
// Column 7 of ones makes the WMMA also produce the denominator sum_r w.
// ---------------------------------------------------------------------------
__global__ void anfis_pack_consequents(const float* __restrict__ cons,
                                       _Float16* __restrict__ bpack) {
    int idx  = blockIdx.x * blockDim.x + threadIdx.x;   // 0 .. 65535
    int e    = idx & 15;
    int lane = (idx >> 4) & 31;
    int k    = idx >> 9;
    int n    = lane & 15;
    int r    = k * 32 + ((e >> 3) << 4) + ((lane >> 4) << 3) + (e & 7);
    float v  = 0.0f;
    if (n < 7)       v = cons[r * 7 + n];
    else if (n == 7) v = 1.0f;
    bpack[idx] = (_Float16)v;
}

static __device__ __forceinline__ h2 splat(_Float16 x) { h2 r = {x, x}; return r; }

static __device__ __forceinline__ v16h mk16(h2 a, h2 b, h2 c, h2 d,
                                            h2 e, h2 f, h2 g, h2 h) {
    v4h p0 = __builtin_shufflevector(a, b, 0, 1, 2, 3);
    v4h p1 = __builtin_shufflevector(c, d, 0, 1, 2, 3);
    v4h p2 = __builtin_shufflevector(e, f, 0, 1, 2, 3);
    v4h p3 = __builtin_shufflevector(g, h, 0, 1, 2, 3);
    v8h q0 = __builtin_shufflevector(p0, p1, 0, 1, 2, 3, 4, 5, 6, 7);
    v8h q1 = __builtin_shufflevector(p2, p3, 0, 1, 2, 3, 4, 5, 6, 7);
    return __builtin_shufflevector(q0, q1, 0, 1, 2, 3, 4, 5, 6, 7,
                                   8, 9, 10, 11, 12, 13, 14, 15);
}

// ---------------------------------------------------------------------------
// Main kernel: 256 threads = 8 waves; each wave computes a 16-row batch tile.
// g[16 x 8] = w_tile @ Cpad over K=4096 rules via 128 v_wmma_f32_16x16x32_f16,
// with the A fragment (rule strengths) synthesized in-register from packed
// f16 pair-tables: w[r] = P01[r>>8] * P23[(r>>4)&15] * P45[r&15].
// ---------------------------------------------------------------------------
__global__ __launch_bounds__(256) void anfis_wmma_kernel(
    const float* __restrict__ x,        // [B,6]
    const float* __restrict__ centers,  // [6,4]
    const float* __restrict__ sigmas,   // [6,4]
    const _Float16* __restrict__ bpack, // packed B fragments (128 chunks)
    float* __restrict__ out)            // [B]
{
    __shared__ float s_acc[8][16][16];  // [wave][M][N]

    const int lane = threadIdx.x & 31;
    const int wid  = threadIdx.x >> 5;
    const int row  = lane & 15;         // batch row within the 16-row tile
    const int hi   = lane >> 4;         // half-wave id
    const int tile = blockIdx.x * 128 + wid * 16;
    const int b    = tile + row;

    // ---- load x row -------------------------------------------------------
    float xv[6];
#pragma unroll
    for (int i = 0; i < 6; ++i) xv[i] = x[b * 6 + i];

    // ---- Gaussian memberships, normalized per input so max_t mu = 1 -------
    // (num/den ratio is invariant to per-row scaling of w; this keeps all
    //  rule strengths representable in f16: the best-matching rule has w == 1)
    float mu[6][4];
#pragma unroll
    for (int i = 0; i < 6; ++i) {
        float mmax = 0.0f;
#pragma unroll
        for (int t = 0; t < 4; ++t) {
            float s = fabsf(sigmas[i * 4 + t]) + 1e-6f;
            float d = xv[i] - centers[i * 4 + t];
            float e = __expf(-(d * d) / (2.0f * s * s));
            mu[i][t] = e;
            mmax = fmaxf(mmax, e);
        }
        float inv = 1.0f / fmaxf(mmax, 1e-30f);
#pragma unroll
        for (int t = 0; t < 4; ++t) mu[i][t] *= inv;
    }

    // ---- packed-f16 pair tables ------------------------------------------
    // P01 as 16 broadcast pairs (indexed by k>>3, static after full unroll)
    h2 a_pk[16];
#pragma unroll
    for (int u = 0; u < 16; ++u)
        a_pk[u] = splat((_Float16)(mu[0][u >> 2] * mu[1][u & 3]));

    // P23 as 8 packed pairs: P23_pk[j] = { P23[2j], P23[2j+1] }
    h2 mu3lo = { (_Float16)mu[3][0], (_Float16)mu[3][1] };
    h2 mu3hi = { (_Float16)mu[3][2], (_Float16)mu[3][3] };
    h2 P23_pk[8];
#pragma unroll
    for (int j = 0; j < 8; ++j)
        P23_pk[j] = splat((_Float16)mu[2][j >> 1]) * ((j & 1) ? mu3hi : mu3lo);

    // local P45 slice (this half-wave's 8 values) as 4 packed pairs
    float m4a = hi ? mu[4][2] : mu[4][0];
    float m4b = hi ? mu[4][3] : mu[4][1];
    h2 mu5lo = { (_Float16)mu[5][0], (_Float16)mu[5][1] };
    h2 mu5hi = { (_Float16)mu[5][2], (_Float16)mu[5][3] };
    h2 p45_pk[4];
    p45_pk[0] = splat((_Float16)m4a) * mu5lo;
    p45_pk[1] = splat((_Float16)m4a) * mu5hi;
    p45_pk[2] = splat((_Float16)m4b) * mu5lo;
    p45_pk[3] = splat((_Float16)m4b) * mu5hi;

    // ---- K loop: 128 chunks of 32 rules, one WMMA each --------------------
    // A fragment layout (16x32 f16): lanes 0-15 K={0..7,16..23},
    // lanes 16-31 K={8..15,24..31}; halves e<8 use c0, e>=8 use c1,
    // P45 index = K&15 -> local p45 pair table for both half-waves.
    const _Float16* bp = bpack + (size_t)lane * 16;

    auto make_af = [&](int k) -> v16h {
        h2 c01 = a_pk[k >> 3] * P23_pk[k & 7];              // {c0, c1}
        h2 c0s = __builtin_shufflevector(c01, c01, 0, 0);
        h2 c1s = __builtin_shufflevector(c01, c01, 1, 1);
        return mk16(c0s * p45_pk[0], c0s * p45_pk[1],
                    c0s * p45_pk[2], c0s * p45_pk[3],
                    c1s * p45_pk[0], c1s * p45_pk[1],
                    c1s * p45_pk[2], c1s * p45_pk[3]);
    };

    v8f acc = {};
    v16h af = make_af(0);                 // software pipeline: build ahead so
#pragma unroll                            // WMMA A-regs aren't rewritten right
    for (int k = 0; k < 128; ++k) {       // after issue (kills hazard v_nops)
        v16h bf = *(const v16h*)(bp + (size_t)k * 512);
        v16h afn = make_af(k + 1 < 128 ? k + 1 : 0);
        acc = __builtin_amdgcn_wmma_f32_16x16x32_f16(
            false, af, false, bf, (short)0, acc, false, false);
        af = afn;
    }

    // ---- stage D (16x16 f32) through LDS ----------------------------------
    // lane L holds column N=L%16, rows M = j (+8 for high half-wave)
#pragma unroll
    for (int j = 0; j < 8; ++j)
        s_acc[wid][hi * 8 + j][row] = acc[j];
    __syncthreads();

    // ---- epilogue: out[b] = (sum_n x_aug*g[:,n]) / (g[:,7] + 1e-8) --------
    if (lane < 16) {
        float num = 0.0f;
#pragma unroll
        for (int n = 0; n < 6; ++n) num += s_acc[wid][row][n] * xv[n];
        num += s_acc[wid][row][6];                // bias column (x_aug last = 1)
        float den = s_acc[wid][row][7] + 1e-8f;   // ones-column of Cpad
        out[b] = num / den;
    }
}

extern "C" void kernel_launch(void* const* d_in, const int* in_sizes, int n_in,
                              void* d_out, int out_size, void* d_ws, size_t ws_size,
                              hipStream_t stream) {
    const float* x       = (const float*)d_in[0];   // [B,6]
    const float* centers = (const float*)d_in[1];   // [6,4]
    const float* sigmas  = (const float*)d_in[2];   // [6,4]
    const float* cons    = (const float*)d_in[3];   // [4096,7]
    float*       out     = (float*)d_out;
    _Float16*    bpack   = (_Float16*)d_ws;         // needs 128 KB

    const int B = in_sizes[0] / 6;                  // 32768

    // 128 chunks * 32 lanes * 16 halves = 65536 f16 elements
    anfis_pack_consequents<<<256, 256, 0, stream>>>(cons, bpack);
    anfis_wmma_kernel<<<B / 128, 256, 0, stream>>>(x, centers, sigmas, bpack, out);
}
